// CellModelxLSTM_83056077570142
// MI455X (gfx1250) — matrix operand order repaired and checked by reading, not compiled, not run on hardware
//
#include <hip/hip_runtime.h>
#include <hip/hip_bf16.h>

// ---------------- model dims ----------------
#define C_NN   207
#define C_OL   3
#define C_D    128
#define C_NB   2
#define C_NH   4
#define C_DI   256
#define C_DH   64
#define C_K    4
#define C_B    64
#define C_S    1024
#define NPOS   (C_B * C_S)        // 65536 token positions

// ---------------- WMMA types ----------------
typedef __attribute__((ext_vector_type(16))) __bf16        v16bf;
typedef __attribute__((ext_vector_type(8)))  float         v8f;
typedef __attribute__((ext_vector_type(8)))  unsigned int  v8u;

union FragBF { v8u u; v16bf bf; };

static __device__ __forceinline__ unsigned short f2bf(float f) {
  union { float f; unsigned u; } c; c.f = f;
  unsigned r = c.u + 0x7FFFu + ((c.u >> 16) & 1u);   // RNE
  return (unsigned short)(r >> 16);
}

static __device__ __forceinline__ unsigned int pk2(float lo, float hi) {
  return (unsigned int)f2bf(lo) | ((unsigned int)f2bf(hi) << 16);
}

static __device__ __forceinline__ float silu(float x) {
  return x / (1.f + expf(-x));
}

// ============================================================
// Weight prep: W[Kd,N] f32 (row-major) -> Wt[Npad][Kpad] bf16,
// zero padded. Run once per launch; tiny (<2 MB total), L2-resident.
// ============================================================
__global__ __launch_bounds__(256)
void wprep_kernel(const float* __restrict__ W, unsigned short* __restrict__ Wt,
                  int Kd, int N, int Kpad, int Npad) {
  size_t idx = (size_t)blockIdx.x * blockDim.x + threadIdx.x;
  if (idx >= (size_t)Npad * Kpad) return;
  int k = (int)(idx % Kpad);
  int n = (int)(idx / Kpad);
  float vv = (k < Kd && n < N) ? W[(size_t)k * N + n] : 0.f;
  Wt[idx] = f2bf(vv);
}

// ============================================================
// WMMA GEMM: C[M,N] = A[M,Kd](lda,f32) * Wt(bf16,[Npad][Kpad]) + bias (+resid)
// 128 threads = 4 waves; 64x64 output tile; wave w owns N-subtile w and all
// four M-subtiles (4 wmma per K-step). A: f32->bf16 staged through double-
// buffered LDS (vectorized fast path / clamped-select ragged path).
// B: loaded directly from pre-transposed bf16 global as 2x b128 per lane.
// ============================================================
#define GT_M 64
#define GT_N 64
#define GT_K 32

static __device__ __forceinline__ v16bf load_frag_a(const unsigned short* row, int hl) {
  // 16-bit A 16x32 layout: VGPR j holds K pair; k = (j>>2)*16 + hl*8 + (j&3)*2
  FragBF f;
  const unsigned int* p0 = (const unsigned int*)(const void*)(row + (hl << 3));
  const unsigned int* p1 = (const unsigned int*)(const void*)(row + 16 + (hl << 3));
#pragma unroll
  for (int j = 0; j < 4; ++j) { f.u[j] = p0[j]; f.u[j + 4] = p1[j]; }
  return f.bf;
}

template <bool RAGGED>
__global__ __launch_bounds__(128)
void wmma_gemm_kernel(const float* __restrict__ A, int lda,
                      const unsigned short* __restrict__ Wt, int Kpad,
                      const float* __restrict__ bias,
                      const float* __restrict__ resid,
                      float* __restrict__ C, int ldc,
                      int M, int N, int Kd) {
  __shared__ __align__(16) unsigned short As[2][GT_M][GT_K];

  const int tid  = threadIdx.x;
  const int lane = tid & 31;
  const int wave = tid >> 5;       // 0..3 : N subtile
  const int hl   = lane >> 4;
  const int l15  = lane & 15;
  const int blockN = blockIdx.x * GT_N;
  const int blockM = blockIdx.y * GT_M;

  const v8f vzero = {};
  v8f acc[4];
#pragma unroll
  for (int i = 0; i < 4; ++i) acc[i] = vzero;

  // ---- A staging helpers ----
  auto stage = [&](int k0, int buf) {
    if (!RAGGED) {
      // lda%4==0, Kd%32==0, M%64==0 guaranteed by dispatcher.
      int r   = tid >> 1;            // 0..63
      int c16 = (tid & 1) * 16;      // 0 or 16
      const float* ap = A + (size_t)(blockM + r) * lda + k0 + c16;
      float4 f0 = *(const float4*)(ap + 0);
      float4 f1 = *(const float4*)(ap + 4);
      float4 f2 = *(const float4*)(ap + 8);
      float4 f3 = *(const float4*)(ap + 12);
      if (k0 + GT_K < Kd) __builtin_prefetch(ap + GT_K, 0, 3);
      unsigned int* dst = (unsigned int*)(void*)&As[buf][r][c16];
      dst[0] = pk2(f0.x, f0.y); dst[1] = pk2(f0.z, f0.w);
      dst[2] = pk2(f1.x, f1.y); dst[3] = pk2(f1.z, f1.w);
      dst[4] = pk2(f2.x, f2.y); dst[5] = pk2(f2.z, f2.w);
      dst[6] = pk2(f3.x, f3.y); dst[7] = pk2(f3.z, f3.w);
    } else {
      // Clamped addresses + select: no divergent exec manipulation.
#pragma unroll
      for (int e = 0; e < 16; ++e) {
        int idx = tid + e * 128;
        int r = idx >> 5, c = idx & 31;
        int gm = blockM + r, gk = k0 + c;
        int gmc = gm < M ? gm : M - 1;
        int gkc = gk < Kd ? gk : Kd - 1;
        float vv = A[(size_t)gmc * lda + gkc];
        As[buf][r][c] = (gm < M && gk < Kd) ? f2bf(vv) : (unsigned short)0;
      }
    }
  };

  const int nk = (Kd + GT_K - 1) / GT_K;
  stage(0, 0);
  __syncthreads();

  const unsigned short* brow0 =
      Wt + (size_t)(blockN + wave * 16 + l15) * Kpad + (hl << 3);

  int buf = 0;
  for (int t = 0; t < nk; ++t) {
    if (t + 1 < nk) stage((t + 1) * GT_K, buf ^ 1);

    // B fragment straight from global: k = hl*8 + 0..7 and 16 + hl*8 + 0..7 per lane-half.
    FragBF fb;
    const uint4* bp = (const uint4*)(const void*)(brow0 + t * GT_K);
    uint4 b0 = bp[0];
    uint4 b1 = bp[2];
    fb.u[0] = b0.x; fb.u[1] = b0.y; fb.u[2] = b0.z; fb.u[3] = b0.w;
    fb.u[4] = b1.x; fb.u[5] = b1.y; fb.u[6] = b1.z; fb.u[7] = b1.w;

#pragma unroll
    for (int tm = 0; tm < 4; ++tm) {
      v16bf am = load_frag_a(&As[buf][tm * 16 + l15][0], hl);
      acc[tm] = __builtin_amdgcn_wmma_f32_16x16x32_bf16(
          false, am, false, fb.bf, (short)0, acc[tm], false, false);
    }
    __syncthreads();
    buf ^= 1;
  }

  // Epilogue: C/D layout lane (hl,l15), VGPR r : row = tm*16 + hl*8 + r
  const int gn = blockN + wave * 16 + l15;
  if (gn >= N) return;
  const float bb = bias ? bias[gn] : 0.f;
#pragma unroll
  for (int tm = 0; tm < 4; ++tm) {
#pragma unroll
    for (int r = 0; r < 8; ++r) {
      int gm = blockM + tm * 16 + hl * 8 + r;
      if (gm < M) {
        size_t off = (size_t)gm * ldc + gn;
        float vv = acc[tm][r] + bb;
        if (resid) vv += resid[off];
        C[off] = vv;
      }
    }
  }
}

// ============================================================
// LayerNorm over 128 features, one wave (32 lanes x 4 elems) per row.
// ============================================================
__global__ __launch_bounds__(256)
void ln_kernel(const float* __restrict__ x, const float* __restrict__ w,
               const float* __restrict__ b, float* __restrict__ out,
               int rows, long ldx) {
  int wid  = (int)(((size_t)blockIdx.x * blockDim.x + threadIdx.x) >> 5);
  int lane = threadIdx.x & 31;
  if (wid >= rows) return;
  const float* xr = x + (size_t)wid * ldx;
  float v0 = xr[lane], v1 = xr[lane + 32], v2 = xr[lane + 64], v3 = xr[lane + 96];
  float s = v0 + v1 + v2 + v3;
  for (int off = 16; off; off >>= 1) s += __shfl_xor(s, off, 32);
  float mu = s * (1.f / 128.f);
  float d0 = v0 - mu, d1 = v1 - mu, d2 = v2 - mu, d3 = v3 - mu;
  float sq = d0 * d0 + d1 * d1 + d2 * d2 + d3 * d3;
  for (int off = 16; off; off >>= 1) sq += __shfl_xor(sq, off, 32);
  float r = rsqrtf(sq * (1.f / 128.f) + 1e-5f);
  float* o = out + (size_t)wid * C_D;
  o[lane]      = d0 * r * w[lane]      + b[lane];
  o[lane + 32] = d1 * r * w[lane + 32] + b[lane + 32];
  o[lane + 64] = d2 * r * w[lane + 64] + b[lane + 64];
  o[lane + 96] = d3 * r * w[lane + 96] + b[lane + 96];
}

// ============================================================
// Causal conv (K=4) over xm = up[..., :DI] (stride 2*DI) + SiLU -> act
// ============================================================
__global__ __launch_bounds__(256)
void conv_silu_kernel(const float* __restrict__ up,
                      const float* __restrict__ w_conv,
                      const float* __restrict__ b_conv,
                      float* __restrict__ act) {
  size_t idx = (size_t)blockIdx.x * blockDim.x + threadIdx.x;
  if (idx >= (size_t)NPOS * C_DI) return;
  int c = (int)(idx & (C_DI - 1));
  size_t p = idx >> 8;              // b*S + s
  int s = (int)(p & (C_S - 1));
  float a = b_conv[c];
#pragma unroll
  for (int j = 0; j < C_K; ++j) {
    int sp = s + j - (C_K - 1);
    if (sp >= 0) a += up[(p + (size_t)(j - (C_K - 1))) * (2 * C_DI) + c] * w_conv[j * C_DI + c];
  }
  act[idx] = silu(a);
}

// ============================================================
// Gate projections: i,f = concat(q,k,v) @ w (768 x 4) + b. One thread per
// (pos, head); adjacent threads share q/k/v rows (broadcast loads).
// ============================================================
__global__ __launch_bounds__(256)
void gates_kernel(const float* __restrict__ q, const float* __restrict__ k,
                  const float* __restrict__ v,
                  const float* __restrict__ wi, const float* __restrict__ bi,
                  const float* __restrict__ wf, const float* __restrict__ bf,
                  float* __restrict__ gi, float* __restrict__ gf) {
  size_t idx = (size_t)blockIdx.x * blockDim.x + threadIdx.x;
  if (idx >= (size_t)NPOS * C_NH) return;
  int h = (int)(idx & 3);
  size_t p = idx >> 2;
  const float* qp = q + p * C_DI;
  const float* kp = k + p * C_DI;
  const float* vp = v + p * C_DI;
  float ai = bi[h], af = bf[h];
  for (int j = 0; j < C_DI; ++j) {
    float qj = qp[j], kj = kp[j], vj = vp[j];
    ai += qj * wi[j * C_NH + h] + kj * wi[(C_DI + j) * C_NH + h] + vj * wi[(2 * C_DI + j) * C_NH + h];
    af += qj * wf[j * C_NH + h] + kj * wf[(C_DI + j) * C_NH + h] + vj * wf[(2 * C_DI + j) * C_NH + h];
  }
  gi[idx] = ai;
  gf[idx] = af;
}

// ============================================================
// mLSTM recurrent scan: one 64-thread workgroup per (b, head).
// Thread t owns output column t: C[k][t] (64 regs). q/k/v/n staged in LDS.
// ============================================================
__global__ __launch_bounds__(64)
void scan_kernel(const float* __restrict__ q, const float* __restrict__ k,
                 const float* __restrict__ v, const float* __restrict__ gi,
                 const float* __restrict__ gf, float* __restrict__ hs) {
  int bh = blockIdx.x;
  int b  = bh >> 2, hh = bh & 3;
  int t  = threadIdx.x;
  __shared__ float qs[C_DH], ks[C_DH], vs[C_DH], ns[C_DH];
  float C[C_DH];
#pragma unroll
  for (int i = 0; i < C_DH; ++i) C[i] = 0.f;
  ns[t] = 0.f;
  float m = 0.f;
  const float scale = 0.125f;  // DH^-0.5
  const size_t gbase = (size_t)b * C_S * C_NH + hh;
  const size_t abase = (size_t)b * C_S * C_DI + hh * C_DH;

  for (int s = 0; s < C_S; ++s) {
    __syncthreads();
    size_t a = abase + (size_t)s * C_DI;
    qs[t] = q[a + t];
    ks[t] = k[a + t];
    vs[t] = v[a + t];
    __syncthreads();

    float it = gi[gbase + (size_t)s * C_NH];
    float ft = gf[gbase + (size_t)s * C_NH];
    float lf = fminf(ft, 0.f) - log1pf(expf(-fabsf(ft)));  // log_sigmoid
    float mn = fmaxf(lf + m, it);
    float i_ = expf(it - mn);
    float f_ = expf(lf + m - mn);
    m = mn;

    float vt  = vs[t];
    float isc = i_ * scale;
    float hacc = 0.f;
#pragma unroll 8
    for (int kk = 0; kk < C_DH; ++kk) {
      C[kk] = f_ * C[kk] + isc * ks[kk] * vt;
      hacc += C[kk] * qs[kk];
    }
    float nt = f_ * ns[t] + isc * ks[t];
    __syncthreads();
    ns[t] = nt;
    __syncthreads();
    float denom = 0.f;
#pragma unroll 8
    for (int kk = 0; kk < C_DH; ++kk) denom += ns[kk] * qs[kk];
    denom = fmaxf(fabsf(denom), 1.f);
    hs[a + t] = hacc / denom;
  }
}

// ============================================================
// GroupNorm over DH per (b,s,head) + skip*act, * silu(z) -> t buffer.
// One wave per group (32 lanes x 2 elems).
// ============================================================
__global__ __launch_bounds__(256)
void gn_skip_kernel(const float* __restrict__ hs, const float* __restrict__ act,
                    const float* __restrict__ up, const float* __restrict__ gn_w,
                    const float* __restrict__ gn_b, const float* __restrict__ skip,
                    float* __restrict__ t) {
  size_t gid = ((size_t)blockIdx.x * blockDim.x + threadIdx.x) >> 5;
  int lane = threadIdx.x & 31;
  if (gid >= (size_t)NPOS * C_NH) return;
  int hh = (int)(gid & 3);
  size_t p = gid >> 2;
  const float* hp = hs + p * C_DI + hh * C_DH;
  float a = hp[lane], b2 = hp[lane + 32];
  float s = a + b2;
  for (int off = 16; off; off >>= 1) s += __shfl_xor(s, off, 32);
  float mu = s * (1.f / 64.f);
  float da = a - mu, db = b2 - mu;
  float sq = da * da + db * db;
  for (int off = 16; off; off >>= 1) sq += __shfl_xor(sq, off, 32);
  float r = rsqrtf(sq * (1.f / 64.f) + 1e-5f);
#pragma unroll
  for (int e = 0; e < 2; ++e) {
    int dh = lane + e * 32;
    int ch = hh * C_DH + dh;
    float d  = e ? db : da;
    float hn = d * r * gn_w[ch] + gn_b[ch];
    float ho = hn + skip[ch] * act[p * C_DI + ch];
    float z  = up[p * (2 * C_DI) + C_DI + ch];
    t[p * C_DI + ch] = ho * silu(z);
  }
}

// ============================================================
// Host side
// ============================================================
struct BlkP {
  const float *ln_w, *ln_b, *w_up, *b_up, *w_conv, *b_conv;
  const float *w_q, *b_q, *w_k, *b_k, *w_v, *b_v;
  const float *w_i, *b_i, *w_f, *b_f, *skip, *gn_w, *gn_b, *w_down, *b_down;
};

static inline void launch_gemm(const float* A, int lda,
                               const unsigned short* Wt, int Kpad,
                               const float* bias, const float* resid,
                               float* C, int ldc, int M, int N, int Kd,
                               hipStream_t stream) {
  dim3 grid((unsigned)((N + GT_N - 1) / GT_N), (unsigned)((M + GT_M - 1) / GT_M));
  const bool ragged = (Kd % GT_K) || (lda % 4) || (M % GT_M);
  if (ragged)
    wmma_gemm_kernel<true><<<grid, 128, 0, stream>>>(A, lda, Wt, Kpad, bias, resid, C, ldc, M, N, Kd);
  else
    wmma_gemm_kernel<false><<<grid, 128, 0, stream>>>(A, lda, Wt, Kpad, bias, resid, C, ldc, M, N, Kd);
}

extern "C" void kernel_launch(void* const* d_in, const int* in_sizes, int n_in,
                              void* d_out, int out_size, void* d_ws, size_t ws_size,
                              hipStream_t stream) {
  // Input flattening: support both jax-pytree sorted-key order
  // (params..., x last -> in_sizes[0]==256==b_conv) and insertion order
  // (x first -> in_sizes[0]==B*S*NN).
  const bool insertion = (in_sizes[0] == C_B * C_S * C_NN);

  const float *x, *in_w, *in_b, *out_w, *out_b, *pln_w, *pln_b;
  BlkP blk[C_NB];

  if (insertion) {
    x     = (const float*)d_in[0];
    in_w  = (const float*)d_in[1];
    in_b  = (const float*)d_in[2];
    for (int i = 0; i < C_NB; ++i) {
      const int o = 3 + i * 21;
      blk[i].ln_w = (const float*)d_in[o + 0];  blk[i].ln_b = (const float*)d_in[o + 1];
      blk[i].w_up = (const float*)d_in[o + 2];  blk[i].b_up = (const float*)d_in[o + 3];
      blk[i].w_conv = (const float*)d_in[o + 4]; blk[i].b_conv = (const float*)d_in[o + 5];
      blk[i].w_q = (const float*)d_in[o + 6];   blk[i].b_q = (const float*)d_in[o + 7];
      blk[i].w_k = (const float*)d_in[o + 8];   blk[i].b_k = (const float*)d_in[o + 9];
      blk[i].w_v = (const float*)d_in[o + 10];  blk[i].b_v = (const float*)d_in[o + 11];
      blk[i].w_i = (const float*)d_in[o + 12];  blk[i].b_i = (const float*)d_in[o + 13];
      blk[i].w_f = (const float*)d_in[o + 14];  blk[i].b_f = (const float*)d_in[o + 15];
      blk[i].skip = (const float*)d_in[o + 16];
      blk[i].gn_w = (const float*)d_in[o + 17]; blk[i].gn_b = (const float*)d_in[o + 18];
      blk[i].w_down = (const float*)d_in[o + 19]; blk[i].b_down = (const float*)d_in[o + 20];
    }
    pln_w = (const float*)d_in[45]; pln_b = (const float*)d_in[46];
    out_w = (const float*)d_in[47]; out_b = (const float*)d_in[48];
  } else {
    // sorted: blocks(2x21 sorted keys), in_b, in_w, out_b, out_w, post_ln_b, post_ln_w, x
    for (int i = 0; i < C_NB; ++i) {
      const int o = i * 21;
      blk[i].b_conv = (const float*)d_in[o + 0];  blk[i].b_down = (const float*)d_in[o + 1];
      blk[i].b_f = (const float*)d_in[o + 2];     blk[i].b_i = (const float*)d_in[o + 3];
      blk[i].b_k = (const float*)d_in[o + 4];     blk[i].b_q = (const float*)d_in[o + 5];
      blk[i].b_up = (const float*)d_in[o + 6];    blk[i].b_v = (const float*)d_in[o + 7];
      blk[i].gn_b = (const float*)d_in[o + 8];    blk[i].gn_w = (const float*)d_in[o + 9];
      blk[i].ln_b = (const float*)d_in[o + 10];   blk[i].ln_w = (const float*)d_in[o + 11];
      blk[i].skip = (const float*)d_in[o + 12];   blk[i].w_conv = (const float*)d_in[o + 13];
      blk[i].w_down = (const float*)d_in[o + 14]; blk[i].w_f = (const float*)d_in[o + 15];
      blk[i].w_i = (const float*)d_in[o + 16];    blk[i].w_k = (const float*)d_in[o + 17];
      blk[i].w_q = (const float*)d_in[o + 18];    blk[i].w_up = (const float*)d_in[o + 19];
      blk[i].w_v = (const float*)d_in[o + 20];
    }
    in_b  = (const float*)d_in[42]; in_w  = (const float*)d_in[43];
    out_b = (const float*)d_in[44]; out_w = (const float*)d_in[45];
    pln_b = (const float*)d_in[46]; pln_w = (const float*)d_in[47];
    x     = (const float*)d_in[48];
  }

  // Workspace layout (floats first, then bf16 weight pool). ~540 MB + 1.4 MB.
  float* ws = (float*)d_ws;
  const size_t P = (size_t)NPOS;
  float* h    = ws;  ws += P * C_D;          // block stream (residual)
  float* xn   = ws;  ws += P * C_D;          // layernorm output
  float* up   = ws;  ws += P * (2 * C_DI);   // up projection (xm | z)
  float* act  = ws;  ws += P * C_DI;         // conv+silu output
  float* qb   = ws;  ws += P * C_DI;         // q (reused as t-buffer post-scan)
  float* kb   = ws;  ws += P * C_DI;         // k
  float* vb   = ws;  ws += P * C_DI;         // v
  float* hsb  = ws;  ws += P * C_DI;         // scan output
  float* gib  = ws;  ws += P * C_NH;         // i gate preact
  float* gfb  = ws;  ws += P * C_NH;         // f gate preact
  float* last = ws;  ws += (size_t)C_B * C_D;
  float* tbuf = qb;                          // alias: q dead after scan

  // bf16 weight pool (16B aligned)
  unsigned short* wpool =
      (unsigned short*)(((uintptr_t)ws + 15) & ~(uintptr_t)15);

  auto prep = [&](const float* W, int Kd, int N, int& Kpad) -> unsigned short* {
    int Kp = (Kd + 31) & ~31;
    int Np = (N + 63) & ~63;
    size_t sz = (size_t)Kp * Np;
    unsigned short* p = wpool;
    wpool += sz;
    wprep_kernel<<<(unsigned)((sz + 255) / 256), 256, 0, stream>>>(W, p, Kd, N, Kp, Np);
    Kpad = Kp;
    return p;
  };

  int kp_in, kp_up[C_NB], kp_q[C_NB], kp_k[C_NB], kp_v[C_NB], kp_dn[C_NB], kp_out;
  unsigned short* wt_in = prep(in_w, C_NN, C_D, kp_in);
  unsigned short *wt_up[C_NB], *wt_q[C_NB], *wt_k[C_NB], *wt_v[C_NB], *wt_dn[C_NB];
  for (int i = 0; i < C_NB; ++i) {
    wt_up[i] = prep(blk[i].w_up,   C_D,  2 * C_DI, kp_up[i]);
    wt_q[i]  = prep(blk[i].w_q,    C_DI, C_DI,     kp_q[i]);
    wt_k[i]  = prep(blk[i].w_k,    C_DI, C_DI,     kp_k[i]);
    wt_v[i]  = prep(blk[i].w_v,    C_DI, C_DI,     kp_v[i]);
    wt_dn[i] = prep(blk[i].w_down, C_DI, C_D,      kp_dn[i]);
  }
  unsigned short* wt_out = prep(out_w, C_D, C_NN * C_OL, kp_out);

  // h = x @ in_w + in_b   (K=207 ragged A-side)
  launch_gemm(x, C_NN, wt_in, kp_in, in_b, nullptr, h, C_D, NPOS, C_D, C_NN, stream);

  for (int i = 0; i < C_NB; ++i) {
    const BlkP& p = blk[i];
    ln_kernel<<<NPOS * 32 / 256, 256, 0, stream>>>(h, p.ln_w, p.ln_b, xn, NPOS, (long)C_D);
    launch_gemm(xn, C_D, wt_up[i], kp_up[i], p.b_up, nullptr, up, 2 * C_DI, NPOS, 2 * C_DI, C_D, stream);
    conv_silu_kernel<<<(unsigned)((size_t)NPOS * C_DI / 256), 256, 0, stream>>>(up, p.w_conv, p.b_conv, act);
    launch_gemm(act, C_DI, wt_q[i], kp_q[i], p.b_q, nullptr, qb, C_DI, NPOS, C_DI, C_DI, stream);
    launch_gemm(act, C_DI, wt_k[i], kp_k[i], p.b_k, nullptr, kb, C_DI, NPOS, C_DI, C_DI, stream);
    launch_gemm(up, 2 * C_DI, wt_v[i], kp_v[i], p.b_v, nullptr, vb, C_DI, NPOS, C_DI, C_DI, stream);
    gates_kernel<<<NPOS * C_NH / 256, 256, 0, stream>>>(qb, kb, vb, p.w_i, p.b_i, p.w_f, p.b_f, gib, gfb);
    scan_kernel<<<C_B * C_NH, C_DH, 0, stream>>>(qb, kb, vb, gib, gfb, hsb);
    gn_skip_kernel<<<NPOS * C_NH * 32 / 256, 256, 0, stream>>>(hsb, act, up, p.gn_w, p.gn_b, p.skip, tbuf);
    // h = h + (t @ w_down + b_down)   (residual fused, in-place)
    launch_gemm(tbuf, C_DI, wt_dn[i], kp_dn[i], p.b_down, h, h, C_D, NPOS, C_D, C_DI, stream);
  }

  // post-LN on last tokens -> last[64][128]
  ln_kernel<<<(C_B * 32 + 255) / 256, 256, 0, stream>>>(
      h + (size_t)(C_S - 1) * C_D, pln_w, pln_b, last, C_B, (long)(C_S * C_D));
  // out = last @ out_w + out_b  (N=621 handled by padded Wt + store guard)
  launch_gemm(last, C_D, wt_out, kp_out, out_b, nullptr,
              (float*)d_out, C_NN * C_OL, C_B, C_NN * C_OL, C_D, stream);
}
